// AttentionLayer_39470749450706
// MI455X (gfx1250) — matrix operand order repaired and checked by reading, not compiled, not run on hardware
//
#include <hip/hip_runtime.h>

typedef __attribute__((ext_vector_type(16))) _Float16 v16h;
typedef __attribute__((ext_vector_type(8)))  float    v8f;
typedef __attribute__((ext_vector_type(4)))  float    v4f;
typedef __attribute__((ext_vector_type(2)))  float    v2f;
typedef __attribute__((ext_vector_type(4)))  int      v4i;

#define GN    8192
#define DIN   256
#define DOUT  128
#define NEGINF (-9.0e15f)
#define ALPHA  0.2f
#define EPS    1e-12f

__device__ __forceinline__ float leaky(float x) { return x >= 0.0f ? x : ALPHA * x; }

// ---------------------------------------------------------------------------
// Kernel 1: h = x @ W  (fp32 WMMA 16x16x4), also emit hT in f16 (transposed,
// so the flash kernel's B-operand loads are contiguous per lane).
// Block = 256 threads = 8 waves. Wave w computes the 16x16 tile at col 16*w
// of the 16-row block blockIdx.x. K loop: 256 / 4 = 64 WMMA steps.
// ---------------------------------------------------------------------------
__global__ __launch_bounds__(256) void gemm_h_kernel(
    const float* __restrict__ x, const float* __restrict__ W,
    float* __restrict__ h, _Float16* __restrict__ hT) {
  const int wave = threadIdx.x >> 5;
  const int lane = threadIdx.x & 31;
  const int row0 = blockIdx.x * 16;
  const int col0 = wave * 16;
  const int m    = lane & 15;
  const int half = lane >> 4;

  v8f c = {};
  // A (16x4 f32): lanes 0-15 hold row M=lane, K = k+{0,1}; lanes 16-31: K = k+2+{0,1}
  const float* xrow = x + (size_t)(row0 + m) * DIN + half * 2;
  // B (4x16 f32): lanes 0-15 hold col N=lane, K = k+{0,1}; lanes 16-31: K = k+2+{0,1}
  const float* wcol = W + col0 + m;

#pragma unroll 4
  for (int k = 0; k < DIN; k += 4) {
    v2f a = *(const v2f*)(xrow + k);
    v2f b;
    b.x = wcol[(size_t)(k + half * 2)     * DOUT];
    b.y = wcol[(size_t)(k + half * 2 + 1) * DOUT];
    c = __builtin_amdgcn_wmma_f32_16x16x4_f32(false, a, false, b, (short)0, c,
                                              false, false);
  }

  // C layout: VGPR r -> row M = r + half*8, col N = m (within tile)
  const int n = col0 + m;
#pragma unroll
  for (int r = 0; r < 8; ++r) {
    const int row = row0 + r + half * 8;
    const float v = c[r];
    h[(size_t)row * DOUT + n] = v;
    hT[(size_t)n * GN + row]  = (_Float16)v;
  }
}

// ---------------------------------------------------------------------------
// Kernel 2: s1 = h . a[:128],  s2 = h . a[128:]   (one row per wave)
// ---------------------------------------------------------------------------
__global__ __launch_bounds__(256) void s12_kernel(
    const float* __restrict__ h, const float* __restrict__ a,
    float* __restrict__ s1, float* __restrict__ s2) {
  const int wave = threadIdx.x >> 5;
  const int lane = threadIdx.x & 31;
  const int row  = blockIdx.x * 8 + wave;
  const float* hr = h + (size_t)row * DOUT;
  float d1 = 0.0f, d2 = 0.0f;
#pragma unroll
  for (int i = 0; i < DOUT; i += 32) {
    const float v = hr[i + lane];
    d1 += v * a[i + lane];
    d2 += v * a[DOUT + i + lane];
  }
#pragma unroll
  for (int off = 16; off > 0; off >>= 1) {
    d1 += __shfl_xor(d1, off, 32);
    d2 += __shfl_xor(d2, off, 32);
  }
  if (lane == 0) { s1[row] = d1; s2[row] = d2; }
}

// ---------------------------------------------------------------------------
// Kernel 3: maxS2 = max_j s2[j]  (single block). Used as a safe upper bound
// for the softmax shift: max_j leaky(s1_i + s2_j) <= leaky(s1_i + maxS2),
// so the per-row max pass over adj (256 MB!) can be skipped entirely.
// ---------------------------------------------------------------------------
__global__ __launch_bounds__(256) void maxs2_kernel(
    const float* __restrict__ s2, float* __restrict__ maxs2) {
  __shared__ float red[8];
  const int lane = threadIdx.x & 31;
  const int wave = threadIdx.x >> 5;
  float mx = NEGINF;
  for (int i = threadIdx.x; i < GN; i += 256) mx = fmaxf(mx, s2[i]);
#pragma unroll
  for (int off = 16; off > 0; off >>= 1) mx = fmaxf(mx, __shfl_xor(mx, off, 32));
  if (lane == 0) red[wave] = mx;
  __syncthreads();
  if (threadIdx.x == 0) {
    float m = red[0];
#pragma unroll
    for (int w = 1; w < 8; ++w) m = fmaxf(m, red[w]);
    *maxs2 = m;
  }
}

// ---------------------------------------------------------------------------
// Kernel 4: fused masked-softmax attention, single pass.
// One block (8 waves) owns a 16-row block; the 8 waves split the 8192-column
// j-loop (wave w takes every 8th 32-column stripe). Each wave builds the
// 16x32 probability tile p = exp(e - B_row) in f16 (B_row = leaky(s1+maxS2)
// upper bound -> all p in [0,1], f16-safe; masked -> exp(-huge) = 0) and
// contracts against hT with 8x v_wmma_f32_16x16x32_f16 per stripe.
// Partial f32 accumulators + denominators are combined once through LDS,
// then the epilogue does 1/l, leaky-relu, row L2-normalize, +bias with
// coalesced float4 stores.
// A-operand lane layout (16-bit A 16x32, same k order as B): lane half h0/h1 ->
// K base = half*16; slots 0-15 hold K=base+0..15 (contiguous), row M = lane&15.
// B-operand: lane half -> K = half*16 + slot (contiguous 16), col N = lane&15.
// ---------------------------------------------------------------------------
__global__ __launch_bounds__(256) void gat_flash_kernel(
    const int* __restrict__ adj, const float* __restrict__ s1,
    const float* __restrict__ s2, const float* __restrict__ maxs2,
    const _Float16* __restrict__ hT, const float* __restrict__ bias,
    float* __restrict__ out) {
  __shared__ float lds_acc[8][16][DOUT];  // 64 KB partial accumulators
  __shared__ float lds_l[8][16];          // partial softmax denominators

  const int wave  = threadIdx.x >> 5;
  const int lane  = threadIdx.x & 31;
  const int row0  = blockIdx.x * 16;
  const int m     = lane & 15;
  const int half  = lane >> 4;
  const int base0 = half * 8;
  const int myrow = row0 + m;

  const float s1r = s1[myrow];
  const float mx  = leaky(s1r + *maxs2);  // upper bound on this row's max score
  const int* adjrow = adj + (size_t)myrow * GN;

  v8f acc[8];
#pragma unroll
  for (int t = 0; t < 8; ++t) acc[t] = (v8f){};
  float l = 0.0f;

  for (int it = 0; it < GN / (32 * 8); ++it) {
    const int j  = (it * 8 + wave) * 32;
    const int kb = j + half * 16;
    const v4i A0 = *(const v4i*)(adjrow + kb);
    const v4i A1 = *(const v4i*)(adjrow + kb + 4);
    const v4i A2 = *(const v4i*)(adjrow + kb + 8);
    const v4i A3 = *(const v4i*)(adjrow + kb + 12);
    const v4f Z0 = *(const v4f*)(s2 + kb);
    const v4f Z1 = *(const v4f*)(s2 + kb + 4);
    const v4f Z2 = *(const v4f*)(s2 + kb + 8);
    const v4f Z3 = *(const v4f*)(s2 + kb + 12);

    v16h av;
#pragma unroll
    for (int q = 0; q < 4; ++q) {
      float e, p0, p1, p2, p3;
      e = leaky(s1r + Z0[q]); e = (A0[q] > 0) ? e : NEGINF; p0 = __expf(e - mx);
      e = leaky(s1r + Z1[q]); e = (A1[q] > 0) ? e : NEGINF; p1 = __expf(e - mx);
      e = leaky(s1r + Z2[q]); e = (A2[q] > 0) ? e : NEGINF; p2 = __expf(e - mx);
      e = leaky(s1r + Z3[q]); e = (A3[q] > 0) ? e : NEGINF; p3 = __expf(e - mx);
      l += (p0 + p1) + (p2 + p3);
      av[q]      = (_Float16)p0;   // K = half*16 + q
      av[4 + q]  = (_Float16)p1;   // K = half*16 + 4 + q
      av[8 + q]  = (_Float16)p2;   // K = half*16 + 8 + q
      av[12 + q] = (_Float16)p3;   // K = half*16 + 12 + q
    }

#pragma unroll
    for (int t = 0; t < 8; ++t) {
      // B tile: h rows j..j+31 (K), cols t*16..t*16+15 (N). hT is [DOUT][GN],
      // so a lane's 16 K-values are one contiguous 32-byte load.
      const _Float16* bp = hT + (size_t)(t * 16 + m) * GN + j + half * 16;
      const v16h bv = *(const v16h*)bp;
      acc[t] = __builtin_amdgcn_wmma_f32_16x16x32_f16(false, av, false, bv,
                                                      (short)0, acc[t],
                                                      false, false);
    }
  }

  // ---- Combine the 8 waves' partials through LDS --------------------------
  l += __shfl_xor(l, 16, 32);        // join the two K-halves of each row
  if (lane < 16) lds_l[wave][m] = l;
#pragma unroll
  for (int t = 0; t < 8; ++t)
#pragma unroll
    for (int r = 0; r < 8; ++r)
      lds_acc[wave][base0 + r][t * 16 + m] = acc[t][r];
  __syncthreads();

  // ---- Epilogue: tid -> (row = tid/16, 8 contiguous cols) -----------------
  const int row = threadIdx.x >> 4;        // 0..15
  const int c0  = (threadIdx.x & 15) * 8;  // col group
  float lrow = 0.0f;
#pragma unroll
  for (int w = 0; w < 8; ++w) lrow += lds_l[w][row];
  const float li = 1.0f / fmaxf(lrow, 1e-30f);

  float v[8];
  float ssq = 0.0f;
#pragma unroll
  for (int q = 0; q < 8; ++q) {
    float s = 0.0f;
#pragma unroll
    for (int w = 0; w < 8; ++w) s += lds_acc[w][row][c0 + q];
    s = leaky(s * li);                 // softmax-normalize then leaky_relu
    v[q] = s;
    ssq += s * s;
  }
  // Row sum-of-squares: the 16 threads of a row sit in one 16-lane half.
  ssq += __shfl_xor(ssq, 1, 32);
  ssq += __shfl_xor(ssq, 2, 32);
  ssq += __shfl_xor(ssq, 4, 32);
  ssq += __shfl_xor(ssq, 8, 32);
  const float ninv = 1.0f / fmaxf(sqrtf(ssq), EPS);

  float* op = out + (size_t)(row0 + row) * DOUT + c0;
  const float* bp = bias + c0;
  v4f o0, o1;
#pragma unroll
  for (int q = 0; q < 4; ++q) o0[q] = v[q] * ninv + bp[q];
#pragma unroll
  for (int q = 0; q < 4; ++q) o1[q] = v[4 + q] * ninv + bp[4 + q];
  *(v4f*)op       = o0;
  *(v4f*)(op + 4) = o1;
}

// ---------------------------------------------------------------------------
extern "C" void kernel_launch(void* const* d_in, const int* in_sizes, int n_in,
                              void* d_out, int out_size, void* d_ws, size_t ws_size,
                              hipStream_t stream) {
  const float* x    = (const float*)d_in[0];  // [8192, 256]
  const int*   adj  = (const int*)  d_in[1];  // [8192, 8192]
  const float* W    = (const float*)d_in[2];  // [256, 128]
  const float* a    = (const float*)d_in[3];  // [256, 1]
  const float* bias = (const float*)d_in[4];  // [128]
  float* out = (float*)d_out;                 // [8192, 128]

  char* ws = (char*)d_ws;
  float*    h   = (float*)ws;                                  // 4 MB
  _Float16* hT  = (_Float16*)(ws + (size_t)4 * 1024 * 1024);   // 2 MB
  float*    s1  = (float*)(ws + (size_t)6 * 1024 * 1024);      // 32 KB
  float*    s2  = (float*)(ws + (size_t)6 * 1024 * 1024 + 32 * 1024);
  float*    ms2 = (float*)(ws + (size_t)6 * 1024 * 1024 + 64 * 1024);

  // h = x@W (+ f16 transpose). 512 row-blocks, 8 col-tiles per block (8 waves).
  gemm_h_kernel<<<GN / 16, 256, 0, stream>>>(x, W, h, hT);
  // s1/s2 dot products: 8 rows per block.
  s12_kernel<<<GN / 8, 256, 0, stream>>>(h, a, s1, s2);
  // global max of s2 (softmax shift bound).
  maxs2_kernel<<<1, 256, 0, stream>>>(s2, ms2);
  // fused attention: one 16-row block per workgroup, 8 waves split the j-loop.
  gat_flash_kernel<<<GN / 16, 256, 0, stream>>>(adj, s1, s2, ms2, hT, bias, out);
}
